// Attention_77309411328551
// MI455X (gfx1250) — compile-verified, matrix operand order checked by reading, not run on hardware
//
#include <hip/hip_runtime.h>
#include <hip/hip_bf16.h>
#include <float.h>

// ---- problem constants (match reference) ----
#define B_      32
#define H_      32
#define KVH_    8
#define G_      4          // query heads per kv head
#define HD_     128
#define BS_     16         // kv cache block size == WMMA tile M
#define MAXKV_  4096
#define BPS_    256        // blocks per sequence
#define SCALE_  0.08838834764831843f

// ---- kernel config ----
#define NW_     4              // waves per workgroup
#define SPLIT_  4              // kv splits (flash-decoding)
#define NSLOT_  (NW_*SPLIT_)   // total kv slots per (b,kh)
#define CHUNK_  32             // kv positions per wave iteration (2 cache blocks)
#define VP_     18             // LDS pitch (uints) per channel row: 16 kv-pairs + 2 pad
#define WSS_    (2*G_ + G_*HD_)  // ws floats per (b,kh,split): m[4], l[4], acc[4][128]

typedef __attribute__((ext_vector_type(16))) __bf16 v16bf;
typedef __attribute__((ext_vector_type(8)))  float  v8f;

__device__ __forceinline__ unsigned int pack_bf16x2(float a, float b) {
    unsigned short ua = __builtin_bit_cast(unsigned short, (__bf16)a);
    unsigned short ub = __builtin_bit_cast(unsigned short, (__bf16)b);
    return (unsigned int)ua | ((unsigned int)ub << 16);
}

__device__ __forceinline__ v16bf pack16(float4 a, float4 b, float4 c, float4 d) {
    v16bf f;
    f[0]=(__bf16)a.x;  f[1]=(__bf16)a.y;  f[2]=(__bf16)a.z;  f[3]=(__bf16)a.w;
    f[4]=(__bf16)b.x;  f[5]=(__bf16)b.y;  f[6]=(__bf16)b.z;  f[7]=(__bf16)b.w;
    f[8]=(__bf16)c.x;  f[9]=(__bf16)c.y;  f[10]=(__bf16)c.z; f[11]=(__bf16)c.w;
    f[12]=(__bf16)d.x; f[13]=(__bf16)d.y; f[14]=(__bf16)d.z; f[15]=(__bf16)d.w;
    return f;
}

// ============================================================================
// Kernel 1: per-(b,kh,split) flash-decode partial.
// grid = (B*KVH, SPLIT), block = NW*32.
// ============================================================================
__global__ __launch_bounds__(NW_*32)
void pa_decode(const float* __restrict__ q,
               const float* __restrict__ kc,
               const float* __restrict__ vc,
               const int*   __restrict__ seqlens,
               const int*   __restrict__ btab,
               float*       __restrict__ ws)
{
    // V chunk staged transposed as bf16: [channel 0..127][kv-pair 0..15], pitch VP_ uints
    __shared__ unsigned int vbuf[NW_][HD_ * VP_];
    __shared__ float accbuf[NW_][G_][HD_];
    __shared__ float mbuf[NW_][G_];
    __shared__ float lbuf[NW_][G_];

    const int bkh  = blockIdx.x;
    const int b    = bkh / KVH_;
    const int kh   = bkh % KVH_;
    const int tid  = threadIdx.x;
    const int wave = tid >> 5;
    const int lane = tid & 31;
    const int lo   = lane & 15;          // row/column within 16
    const int hi   = (lane >> 4) & 1;    // lane half
    const int slot = blockIdx.y * NW_ + wave;

    const int seqlen  = seqlens[b];
    const int nchunks = (seqlen + CHUNK_ - 1) / CHUNK_;

    // ---- Q fragments: B-operand of S^T = K x Q^T. Column n = head = lo.
    // Per WMMA step j (channels 32j..32j+31), lane needs channels
    // 32j + 8*hi + {0..7} (elems 0..7) and 32j + 8*hi + {16..23} (elems 8..15).
    v16bf qf[4];
    {
        const bool valid = (lo < G_);
        const float* qp = q + (size_t)(b*H_ + kh*G_ + (valid ? lo : 0)) * HD_ + 8*hi;
        #pragma unroll
        for (int j = 0; j < 4; ++j) {
            float4 x0 = *(const float4*)(qp + 32*j);
            float4 x1 = *(const float4*)(qp + 32*j + 4);
            float4 y0 = *(const float4*)(qp + 32*j + 16);
            float4 y1 = *(const float4*)(qp + 32*j + 20);
            if (!valid) { x0 = x1 = y0 = y1 = make_float4(0.f,0.f,0.f,0.f); }
            qf[j] = pack16(x0, x1, y0, y1);
        }
    }

    // running softmax state: per lane, head = lo (duplicated in both halves)
    float m = -FLT_MAX, l = 0.f;
    // PV accumulators: D rows = heads (VGPR r -> head r + 8*hi), cols = channel 16j+lo
    v8f acc[8];
    #pragma unroll
    for (int j = 0; j < 8; ++j)
        #pragma unroll
        for (int r = 0; r < 8; ++r) acc[j][r] = 0.f;

    for (int c = slot; c < nchunks; c += NSLOT_) {
        const int kv0  = c * CHUNK_;
        const int blk0 = btab[b*BPS_ + 2*c];
        const int blk1 = btab[b*BPS_ + 2*c + 1];

        // ---- prefetch next iteration's K/V chunk into L2/WGP$ (one full
        // iteration of latency distance; 32KB, 128B-granular, spread over lanes)
        const int cn = c + NSLOT_;
        if (cn < nchunks) {
            const int nblk0 = btab[b*BPS_ + 2*cn];
            const int nblk1 = btab[b*BPS_ + 2*cn + 1];
            // K tile 0/1: 16 rows x 512B; lane -> row (lo), half picks tile, 4 lines
            {
                const int blk = hi ? nblk1 : nblk0;
                const char* kp = (const char*)(kc + (size_t)((blk*BS_ + lo)*KVH_ + kh) * HD_);
                __builtin_prefetch(kp,       0, 1);
                __builtin_prefetch(kp + 128, 0, 1);
                __builtin_prefetch(kp + 256, 0, 1);
                __builtin_prefetch(kp + 384, 0, 1);
            }
            // V chunk: 32 rows x 512B; lane -> row (lo), half picks block, 4 lines
            {
                const int blk = hi ? nblk1 : nblk0;
                const char* vp = (const char*)(vc + (size_t)((blk*BS_ + lo)*KVH_ + kh) * HD_);
                __builtin_prefetch(vp,       0, 1);
                __builtin_prefetch(vp + 128, 0, 1);
                __builtin_prefetch(vp + 256, 0, 1);
                __builtin_prefetch(vp + 384, 0, 1);
            }
        }

        // ---- S^T = K x Q^T for two 16-kv tiles (rows = kv, cols = heads)
        v8f S[2];
        #pragma unroll
        for (int t = 0; t < 2; ++t) {
            const int blk = t ? blk1 : blk0;
            const float* kp = kc + (size_t)((blk*BS_ + lo)*KVH_ + kh) * HD_ + 8*hi;
            v8f s;
            #pragma unroll
            for (int r = 0; r < 8; ++r) s[r] = 0.f;
            #pragma unroll
            for (int j = 0; j < 4; ++j) {
                float4 x0 = *(const float4*)(kp + 32*j);
                float4 x1 = *(const float4*)(kp + 32*j + 4);
                float4 y0 = *(const float4*)(kp + 32*j + 16);
                float4 y1 = *(const float4*)(kp + 32*j + 20);
                v16bf kf = pack16(x0, x1, y0, y1);
                s = __builtin_amdgcn_wmma_f32_16x16x32_bf16(
                        false, kf, false, qf[j], (short)0, s, false, false);
            }
            // scale + length mask (kv position = tile row = r + 8*hi)
            #pragma unroll
            for (int r = 0; r < 8; ++r) {
                const int sg = kv0 + t*BS_ + r + 8*hi;
                s[r] = (sg < seqlen) ? s[r] * SCALE_ : -FLT_MAX;
            }
            S[t] = s;
        }

        // ---- stage V chunk (32 kv x 128 ch) transposed bf16 into LDS.
        // pass p: all lanes read rows 2p,2p+1; lane covers channels 4*lane..4*lane+3
        // -> each global_load float4 is a fully-coalesced 512B wave access.
        __builtin_amdgcn_fence(__ATOMIC_ACQ_REL, "wavefront");
        #pragma unroll
        for (int p = 0; p < 16; ++p) {
            const int srow = 2*p;
            const int blk  = (srow < BS_) ? blk0 : blk1;
            const int sin  = srow & (BS_ - 1);
            const float* vp0 = vc + (size_t)((blk*BS_ + sin)*KVH_ + kh) * HD_ + 4*lane;
            const float* vp1 = vp0 + (size_t)KVH_ * HD_;   // srow+1, same block (srow even)
            const float4 a  = *(const float4*)vp0;
            const float4 bb = *(const float4*)vp1;
            unsigned int* dst = &vbuf[wave][(4*lane)*VP_ + p];
            dst[0*VP_] = pack_bf16x2(a.x, bb.x);
            dst[1*VP_] = pack_bf16x2(a.y, bb.y);
            dst[2*VP_] = pack_bf16x2(a.z, bb.z);
            dst[3*VP_] = pack_bf16x2(a.w, bb.w);
        }

        // ---- online softmax, per head = per lane column
        float tmax = -FLT_MAX;
        #pragma unroll
        for (int r = 0; r < 8; ++r) tmax = fmaxf(tmax, fmaxf(S[0][r], S[1][r]));
        tmax = fmaxf(tmax, __shfl_xor(tmax, 16, 32));
        const float mnew = fmaxf(m, tmax);
        const float corr = __expf(m - mnew);
        float p0[8], p1[8], psum = 0.f;
        #pragma unroll
        for (int r = 0; r < 8; ++r) {
            p0[r] = __expf(S[0][r] - mnew);
            p1[r] = __expf(S[1][r] - mnew);
            psum += p0[r] + p1[r];
        }
        psum += __shfl_xor(psum, 16, 32);
        l = l * corr + psum;
        m = mnew;

        // P -> A-fragment of PV: pure element-wise (no cross-lane movement):
        // lane row = head = lo; k-slots e<8 -> tile0 kv (r+8*hi), e>=8 -> tile1.
        v16bf pf;
        #pragma unroll
        for (int r = 0; r < 8; ++r) { pf[r] = (__bf16)p0[r]; pf[8+r] = (__bf16)p1[r]; }

        // broadcast per-head rescale factors into accumulator layout
        float corrs[8];
        #pragma unroll
        for (int r = 0; r < 8; ++r) corrs[r] = __shfl(corr, r + 8*hi, 32);

        __builtin_amdgcn_fence(__ATOMIC_ACQ_REL, "wavefront"); // LDS store -> load order

        #pragma unroll
        for (int j = 0; j < 8; ++j) {
            // V B-fragment: column n = channel 16j+lo; kv pairs from transposed LDS
            const unsigned int* src = &vbuf[wave][(16*j + lo)*VP_ + 4*hi];
            v16bf vf;
            #pragma unroll
            for (int t4 = 0; t4 < 4; ++t4) {
                const unsigned int u0 = src[t4];       // kv 8*hi + {2t4, 2t4+1}
                const unsigned int u1 = src[8 + t4];   // kv 16 + 8*hi + {2t4, 2t4+1}
                vf[2*t4]     = __builtin_bit_cast(__bf16, (unsigned short)(u0 & 0xffffu));
                vf[2*t4+1]   = __builtin_bit_cast(__bf16, (unsigned short)(u0 >> 16));
                vf[8+2*t4]   = __builtin_bit_cast(__bf16, (unsigned short)(u1 & 0xffffu));
                vf[8+2*t4+1] = __builtin_bit_cast(__bf16, (unsigned short)(u1 >> 16));
            }
            v8f a = acc[j];
            #pragma unroll
            for (int r = 0; r < 8; ++r) a[r] *= corrs[r];
            acc[j] = __builtin_amdgcn_wmma_f32_16x16x32_bf16(
                         false, pf, false, vf, (short)0, a, false, false);
        }
    }

    // ---- intra-block combine across NW waves -> ws[(bkh, split)]
    if (hi == 0 && lo < G_) { mbuf[wave][lo] = m; lbuf[wave][lo] = l; }
    if (lane < 16) {
        #pragma unroll
        for (int j = 0; j < 8; ++j)
            #pragma unroll
            for (int r = 0; r < G_; ++r)
                accbuf[wave][r][16*j + lane] = acc[j][r];
    }
    __syncthreads();

    const size_t base = ((size_t)bkh * SPLIT_ + blockIdx.y) * WSS_;
    for (int item = tid; item < G_*HD_; item += NW_*32) {
        const int g = item >> 7;           // /HD_
        const int d = item & (HD_ - 1);
        float ms = -FLT_MAX;
        #pragma unroll
        for (int w = 0; w < NW_; ++w) ms = fmaxf(ms, mbuf[w][g]);
        float den = 0.f, num = 0.f;
        #pragma unroll
        for (int w = 0; w < NW_; ++w) {
            const float e = __expf(mbuf[w][g] - ms);
            den += e * lbuf[w][g];
            num += e * accbuf[w][g][d];
        }
        ws[base + 2*G_ + g*HD_ + d] = num;
        if (d == 0) { ws[base + g] = ms; ws[base + G_ + g] = den; }
    }
}

// ============================================================================
// Kernel 2: merge SPLIT partials, normalize, write out. grid = B*KVH, block 128.
// ============================================================================
__global__ __launch_bounds__(128)
void pa_combine(const float* __restrict__ ws, float* __restrict__ out)
{
    const int bkh = blockIdx.x;
    const int b   = bkh / KVH_;
    const int kh  = bkh % KVH_;
    for (int item = threadIdx.x; item < G_*HD_; item += 128) {
        const int g = item >> 7;
        const int d = item & (HD_ - 1);
        float ms = -FLT_MAX;
        #pragma unroll
        for (int s = 0; s < SPLIT_; ++s)
            ms = fmaxf(ms, ws[((size_t)bkh*SPLIT_ + s)*WSS_ + g]);
        float num = 0.f, den = 0.f;
        #pragma unroll
        for (int s = 0; s < SPLIT_; ++s) {
            const size_t base = ((size_t)bkh*SPLIT_ + s)*WSS_;
            const float e = __expf(ws[base + g] - ms);
            den += e * ws[base + G_ + g];
            num += e * ws[base + 2*G_ + g*HD_ + d];
        }
        out[((size_t)(b*H_ + kh*G_ + g))*HD_ + d] = num / den;
    }
}

extern "C" void kernel_launch(void* const* d_in, const int* in_sizes, int n_in,
                              void* d_out, int out_size, void* d_ws, size_t ws_size,
                              hipStream_t stream)
{
    (void)in_sizes; (void)n_in; (void)out_size; (void)ws_size;
    const float* q  = (const float*)d_in[0];
    const float* kc = (const float*)d_in[1];
    const float* vc = (const float*)d_in[2];
    const int*   sl = (const int*)d_in[3];
    const int*   bt = (const int*)d_in[4];
    float* ws = (float*)d_ws;

    dim3 g1(B_*KVH_, SPLIT_);
    pa_decode<<<g1, NW_*32, 0, stream>>>(q, kc, vc, sl, bt, ws);
    pa_combine<<<B_*KVH_, 128, 0, stream>>>(ws, (float*)d_out);
}
